// SleepBlock_16458314678779
// MI455X (gfx1250) — matrix operand order repaired and checked by reading, not compiled, not run on hardware
//
#include <hip/hip_runtime.h>
#include <hip/hip_bf16.h>
#include <stdint.h>

// Problem geometry (fixed by the reference).
#define N_ROWS 16384
#define D_IN   2048
#define D_OUT  2048

typedef __bf16 bf16x16 __attribute__((ext_vector_type(16)));
typedef float  f32x8   __attribute__((ext_vector_type(8)));

struct alignas(16) U4 { uint32_t x, y, z, w; };
union Frag { bf16x16 v; U4 q[2]; };   // 32 bytes per lane = one 16x32 (A) or 32x16 (B) bf16 fragment

// ---------- helpers ----------
static __device__ __forceinline__ unsigned short f32_to_bf16(float f) {
    // round-to-nearest-even
    uint32_t u = __float_as_uint(f);
    uint32_t r = (u + 0x7FFFu + ((u >> 16) & 1u)) >> 16;
    return (unsigned short)r;
}

// gfx1250 async global->LDS copy of 16 bytes per lane (VDST = LDS byte addr, tracked by ASYNCcnt)
static __device__ __forceinline__ void async_copy_b128(uint32_t lds_addr, const void* gaddr) {
    asm volatile("global_load_async_to_lds_b128 %0, %1, off"
                 :: "v"(lds_addr), "v"(gaddr) : "memory");
}
static __device__ __forceinline__ void wait_async_le2() {
    asm volatile("s_wait_asynccnt 0x2" ::: "memory");
}
static __device__ __forceinline__ void wait_async_0() {
    asm volatile("s_wait_asynccnt 0x0" ::: "memory");
}

// ---------- 1a. partial sum of |W| ----------
__global__ __launch_bounds__(256) void abs_sum_partial(const float* __restrict__ W,
                                                       float* __restrict__ part) {
    const int tid = blockIdx.x * 256 + threadIdx.x;
    const int total = D_OUT * D_IN;
    float s = 0.f;
    for (int i = tid; i < total; i += 256 * 256) s += fabsf(W[i]);
    __shared__ float red[256];
    red[threadIdx.x] = s;
    __syncthreads();
    for (int off = 128; off > 0; off >>= 1) {
        if (threadIdx.x < off) red[threadIdx.x] += red[threadIdx.x + off];
        __syncthreads();
    }
    if (threadIdx.x == 0) part[blockIdx.x] = red[0];
}

// ---------- 1b. final: scale = max(mean(|W|), 1e-8) ----------
__global__ __launch_bounds__(256) void abs_sum_final(const float* __restrict__ part,
                                                     float* __restrict__ scale_out) {
    __shared__ float red[256];
    red[threadIdx.x] = part[threadIdx.x];
    __syncthreads();
    for (int off = 128; off > 0; off >>= 1) {
        if (threadIdx.x < off) red[threadIdx.x] += red[threadIdx.x + off];
        __syncthreads();
    }
    if (threadIdx.x == 0)
        scale_out[0] = fmaxf(red[0] / (float)(D_OUT * D_IN), 1e-8f);
}

// ---------- 2. ternary-quantize W and cast We, both written transposed K-major bf16 ----------
// WqT[k][n] = ternary(W[n][k]) * scale ;  WeT[k][n] = bf16(We[n][k])
__global__ __launch_bounds__(256) void quant_transpose(const float* __restrict__ W,
                                                       const float* __restrict__ We,
                                                       const float* __restrict__ scale_p,
                                                       unsigned short* __restrict__ WqT,
                                                       unsigned short* __restrict__ WeT) {
    __shared__ float tw[32][33];
    __shared__ float te[32][33];
    const int tx = threadIdx.x;       // 0..31
    const int ty = threadIdx.y;       // 0..7
    const int k0 = blockIdx.x * 32;
    const int n0 = blockIdx.y * 32;
    const float scale = scale_p[0];
    const float thr = 0.3f * scale;   // ws > 0.3  <=>  W > 0.3*scale (scale > 0)

#pragma unroll
    for (int i = 0; i < 4; ++i) {
        const int nl = ty + 8 * i;
        tw[nl][tx] = W [(size_t)(n0 + nl) * D_IN + k0 + tx];
        te[nl][tx] = We[(size_t)(n0 + nl) * D_IN + k0 + tx];
    }
    __syncthreads();
#pragma unroll
    for (int i = 0; i < 4; ++i) {
        const int kl = ty + 8 * i;
        const float w = tw[tx][kl];
        const float q = (w > thr) ? scale : ((w < -thr) ? -scale : 0.0f);
        WqT[(size_t)(k0 + kl) * D_OUT + n0 + tx] = f32_to_bf16(q);
        WeT[(size_t)(k0 + kl) * D_OUT + n0 + tx] = f32_to_bf16(te[tx][kl]);
    }
}

// ---------- 3. cast x -> bf16 row-major ----------
__global__ __launch_bounds__(256) void cast_x_bf16(const float* __restrict__ x,
                                                   unsigned short* __restrict__ xb) {
    const int i = blockIdx.x * 256 + threadIdx.x;       // one float4 per thread
    const float4 f = ((const float4*)x)[i];
    uint2 o;
    o.x = (uint32_t)f32_to_bf16(f.x) | ((uint32_t)f32_to_bf16(f.y) << 16);
    o.y = (uint32_t)f32_to_bf16(f.z) | ((uint32_t)f32_to_bf16(f.w) << 16);
    ((uint2*)xb)[i] = o;
}

// ---------- 4. fused dual GEMM: y = relu(x@WqT + b) + (x@WeT + be) ----------
// Block: 256 threads = 8 waves (4 M x 2 N); block tile 128x64.
// B tiles (both matrices) double-buffered in LDS via async global->LDS copies (ASYNCcnt).
__global__ __launch_bounds__(256) void gemm_fused(const unsigned short* __restrict__ xb,
                                                  const unsigned short* __restrict__ WqT,
                                                  const unsigned short* __restrict__ WeT,
                                                  const float* __restrict__ b,
                                                  const float* __restrict__ be,
                                                  float* __restrict__ y) {
    // [buf][matrix][k*64 + n] : 2*2*32*64 bf16 = 16 KB
    __shared__ __align__(16) unsigned short smem[2][2][32 * 64];

    const int lane = threadIdx.x & 31;
    const int wave = threadIdx.x >> 5;
    const int wm = wave & 3;          // 0..3 (M)
    const int wn = wave >> 2;         // 0..1 (N)
    const int l16 = lane & 15;
    const int hi  = lane >> 4;

    const int n_blk  = blockIdx.x * 64;
    const int m_base = blockIdx.y * 128 + wm * 32;
    const int n_base = n_blk + wn * 32;

    // async-copy assignment: 256 threads x 16B = one 32x64 bf16 tile per matrix per issue
    const int crow = threadIdx.x >> 3;        // 0..31  (K row within tile)
    const int ccol = (threadIdx.x & 7) * 8;   // 0..56  (N start, 8 halves = 16B)

    const f32x8 z = {0.f, 0.f, 0.f, 0.f, 0.f, 0.f, 0.f, 0.f};
    f32x8 acc1[2][2], acc2[2][2];
#pragma unroll
    for (int mi = 0; mi < 2; ++mi)
#pragma unroll
        for (int ni = 0; ni < 2; ++ni) { acc1[mi][ni] = z; acc2[mi][ni] = z; }

    // A: row-major bf16; lane l<16 holds row m_base+l, K chunks [k0..k0+7],[k0+16..k0+23];
    //    lanes 16-31 hold K chunks [k0+8..15],[k0+24..31]  (ISA 16-bit A layout)
    const unsigned short* arow0 = xb + (size_t)(m_base      + l16) * D_IN;
    const unsigned short* arow1 = xb + (size_t)(m_base + 16 + l16) * D_IN;
    const int ka = hi * 8;

    // issue async B-tile copies for k-step `ks` into buffer `buf`
    auto issue_tile = [&](int ks, int buf) {
        const size_t grow = (size_t)(ks * 32 + crow) * D_OUT + n_blk + ccol;
        async_copy_b128((uint32_t)(uintptr_t)&smem[buf][0][crow * 64 + ccol], WqT + grow);
        async_copy_b128((uint32_t)(uintptr_t)&smem[buf][1][crow * 64 + ccol], WeT + grow);
    };

    issue_tile(0, 0);                                  // prologue
    for (int it = 0; it < D_IN / 32; ++it) {
        const int cur = it & 1;
        const int k0  = it * 32;

        if (it + 1 < D_IN / 32) {
            issue_tile(it + 1, cur ^ 1);               // overlap next tile's copy
            wait_async_le2();                          // retire current buffer's 2 copies
        } else {
            wait_async_0();
        }
        __syncthreads();                               // whole B tile visible to all waves

        Frag a[2];
        a[0].q[0] = *(const U4*)(arow0 + k0      + ka);
        a[0].q[1] = *(const U4*)(arow0 + k0 + 16 + ka);
        a[1].q[0] = *(const U4*)(arow1 + k0      + ka);
        a[1].q[1] = *(const U4*)(arow1 + k0 + 16 + ka);
        if (k0 + 32 < D_IN) {                          // prefetch next A chunks
            __builtin_prefetch((const void*)(arow0 + k0 + 32 + ka), 0, 1);
            __builtin_prefetch((const void*)(arow1 + k0 + 32 + ka), 0, 1);
        }

        // B from LDS: lane = N column (l%16), elements over K in table order (8h+s, 16+8h+s-8)
        const unsigned short* bqb = &smem[cur][0][wn * 32 + l16];
        const unsigned short* beb = &smem[cur][1][wn * 32 + l16];
        Frag bq[2], bf[2];
#pragma unroll
        for (int ni = 0; ni < 2; ++ni)
#pragma unroll
            for (int s = 0; s < 16; ++s) {
                const int kk = (s < 8) ? (hi * 8 + s) : (16 + hi * 8 + (s - 8));
                bq[ni].v[s] = __builtin_bit_cast(__bf16, bqb[kk * 64 + ni * 16]);
                bf[ni].v[s] = __builtin_bit_cast(__bf16, beb[kk * 64 + ni * 16]);
            }

#pragma unroll
        for (int mi = 0; mi < 2; ++mi)
#pragma unroll
            for (int ni = 0; ni < 2; ++ni) {
                acc1[mi][ni] = __builtin_amdgcn_wmma_f32_16x16x32_bf16(
                    false, a[mi].v, false, bq[ni].v, (short)0, acc1[mi][ni], false, false);
                acc2[mi][ni] = __builtin_amdgcn_wmma_f32_16x16x32_bf16(
                    false, a[mi].v, false, bf[ni].v, (short)0, acc2[mi][ni], false, false);
            }

        __syncthreads();   // all waves done reading buf[cur] before it is re-filled
    }

    // Epilogue. C/D layout: vgpr r, lane l -> (M = r + 8*(l>=16), N = l%16)
#pragma unroll
    for (int mi = 0; mi < 2; ++mi)
#pragma unroll
        for (int ni = 0; ni < 2; ++ni) {
            const int col = n_base + ni * 16 + l16;
            const float bb  = b[col];
            const float bbe = be[col];
#pragma unroll
            for (int r = 0; r < 8; ++r) {
                const int row = m_base + mi * 16 + hi * 8 + r;
                float c = acc1[mi][ni][r] + bb;
                c = c > 0.f ? c : 0.f;
                y[(size_t)row * D_OUT + col] = c + acc2[mi][ni][r] + bbe;
            }
        }
}

// ---------- 5. per-row int8 fake quant, in place ----------
__global__ __launch_bounds__(256) void row_quant(float* __restrict__ y) {
    float* p = y + (size_t)blockIdx.x * D_OUT;
    float m = 0.f;
    for (int c = threadIdx.x; c < D_OUT; c += 256) m = fmaxf(m, fabsf(p[c]));
    __shared__ float red[256];
    red[threadIdx.x] = m;
    __syncthreads();
    for (int off = 128; off > 0; off >>= 1) {
        if (threadIdx.x < off) red[threadIdx.x] = fmaxf(red[threadIdx.x], red[threadIdx.x + off]);
        __syncthreads();
    }
    const float mx = fmaxf(red[0], 1e-5f);
    const float s  = 127.0f / mx;
    const float inv = mx / 127.0f;
    for (int c = threadIdx.x; c < D_OUT; c += 256) {
        float v = p[c] * s;
        v = fminf(fmaxf(v, -127.0f), 127.0f);
        p[c] = rintf(v) * inv;     // rintf = round-half-even, matches jnp.round
    }
}

// ---------- workspace layout ----------
static const size_t XB_OFF    = 0;                       // 16384*2048*2  = 67,108,864 B
static const size_t WQT_OFF   = 67108864;                // 2048*2048*2  =  8,388,608 B
static const size_t WET_OFF   = WQT_OFF + 8388608;
static const size_t PART_OFF  = WET_OFF + 8388608;       // 256 floats
static const size_t SCALE_OFF = PART_OFF + 1024;         // 1 float

extern "C" void kernel_launch(void* const* d_in, const int* in_sizes, int n_in,
                              void* d_out, int out_size, void* d_ws, size_t ws_size,
                              hipStream_t stream) {
    const float* x  = (const float*)d_in[0];
    const float* W  = (const float*)d_in[1];
    const float* b  = (const float*)d_in[2];
    const float* We = (const float*)d_in[3];
    const float* be = (const float*)d_in[4];
    float* y = (float*)d_out;

    char* ws = (char*)d_ws;
    unsigned short* xb    = (unsigned short*)(ws + XB_OFF);
    unsigned short* WqT   = (unsigned short*)(ws + WQT_OFF);
    unsigned short* WeT   = (unsigned short*)(ws + WET_OFF);
    float*          part  = (float*)(ws + PART_OFF);
    float*          scale = (float*)(ws + SCALE_OFF);

    abs_sum_partial<<<256, 256, 0, stream>>>(W, part);
    abs_sum_final<<<1, 256, 0, stream>>>(part, scale);
    quant_transpose<<<dim3(D_IN / 32, D_OUT / 32), dim3(32, 8), 0, stream>>>(W, We, scale, WqT, WeT);
    cast_x_bf16<<<(N_ROWS * D_IN / 4) / 256, 256, 0, stream>>>(x, xb);
    gemm_fused<<<dim3(D_OUT / 64, N_ROWS / 128), 256, 0, stream>>>(xb, WqT, WeT, b, be, y);
    row_quant<<<N_ROWS, 256, 0, stream>>>(y);
}
